// MultiHeadSelfAttention_20444044329775
// MI455X (gfx1250) — matrix operand order repaired and checked by reading, not compiled, not run on hardware
//
#include <hip/hip_runtime.h>
#include <stdint.h>

typedef __attribute__((ext_vector_type(16))) __bf16 v16bf;
typedef __attribute__((ext_vector_type(8)))  __bf16 v8bf;
typedef __attribute__((ext_vector_type(4)))  __bf16 v4bf;
typedef __attribute__((ext_vector_type(8)))  float  v8f;
typedef __attribute__((ext_vector_type(4)))  int    v4i;

#define GLOBAL_AS __attribute__((address_space(1)))
#define LDS_AS    __attribute__((address_space(3)))

#if defined(__has_builtin)
#if __has_builtin(__builtin_amdgcn_global_load_async_to_lds_b128)
#define HAVE_ASYNC_LDS 1
#endif
#endif
#ifndef HAVE_ASYNC_LDS
#define HAVE_ASYNC_LDS 0
#endif

// ---------------------------------------------------------------- helpers
__device__ __forceinline__ v8f wmma_bf16(v16bf a, v16bf b, v8f c) {
  return __builtin_amdgcn_wmma_f32_16x16x32_bf16(
      /*neg_a=*/false, a, /*neg_b=*/false, b,
      /*c_mod=*/(short)0, c, /*reuse_a=*/false, /*reuse_b=*/false);
}

// A-fragment style: elements 0..7 from p0, 8..15 from p1 (two 16B chunks)
__device__ __forceinline__ v16bf load2x8(const __bf16* p0, const __bf16* p1) {
  v8bf lo = *(const v8bf*)p0;
  v8bf hi = *(const v8bf*)p1;
  v16bf r;
#pragma unroll
  for (int e = 0; e < 8; ++e) { r[e] = lo[e]; r[e + 8] = hi[e]; }
  return r;
}

// B-fragment style: 16 contiguous bf16
__device__ __forceinline__ v16bf load16(const __bf16* p) {
  return load2x8(p, p + 8);
}

// 16B global -> LDS copy: async on CDNA5 (ASYNCcnt-tracked, bypasses VGPRs)
__device__ __forceinline__ void cp16_g2l(const __bf16* g, __bf16* l) {
#if HAVE_ASYNC_LDS
  __builtin_amdgcn_global_load_async_to_lds_b128(
      (GLOBAL_AS v4i*)(g), (LDS_AS v4i*)(l), 0, 0);
#else
  *(v8bf*)l = *(const v8bf*)g;
#endif
}

__device__ __forceinline__ void wait_async_all() {
#if HAVE_ASYNC_LDS
  asm volatile("s_wait_asynccnt 0" ::: "memory");
#endif
}

// ---------------------------------------------------------------- fp32 -> bf16 for x
__global__ __launch_bounds__(256) void convert_x_kernel(const float* __restrict__ x,
                                                        __bf16* __restrict__ xb) {
  int i = (blockIdx.x * 256 + threadIdx.x) * 4;
  float4 v = *(const float4*)(x + i);
  v4bf o = { (__bf16)v.x, (__bf16)v.y, (__bf16)v.z, (__bf16)v.w };
  *(v4bf*)(xb + i) = o;
}

// ---------------------------------------------------------------- W[k][n] -> Wt[n][k] bf16
__global__ __launch_bounds__(1024) void transpose_w_kernel(
    const float* __restrict__ w0, const float* __restrict__ w1,
    const float* __restrict__ w2, const float* __restrict__ w3,
    __bf16* __restrict__ t0, __bf16* __restrict__ t1,
    __bf16* __restrict__ t2, __bf16* __restrict__ t3) {
  __shared__ float tile[32][33];
  const float* w = (blockIdx.z == 0) ? w0 : (blockIdx.z == 1) ? w1 : (blockIdx.z == 2) ? w2 : w3;
  __bf16*      t = (blockIdx.z == 0) ? t0 : (blockIdx.z == 1) ? t1 : (blockIdx.z == 2) ? t2 : t3;
  int bx = blockIdx.x * 32, by = blockIdx.y * 32;
  tile[threadIdx.y][threadIdx.x] = w[(size_t)(by + threadIdx.y) * 1024 + bx + threadIdx.x];
  __syncthreads();
  t[(size_t)(bx + threadIdx.y) * 1024 + by + threadIdx.x] = (__bf16)tile[threadIdx.x][threadIdx.y];
}

// ---------------------------------------------------------------- GEMM: C = A(bf16)[4096x1024] @ Wt(bf16 transposed)[1024x1024] + bias
// Double-buffered, async global->LDS pipeline:
//   wait(asynccnt) ; barrier ; compute buf[k&1] ; issue async copies into buf[(k+1)&1]
// mode 0: write bf16 at [B,H,S,64]   (Q / K layout)
// mode 1: write bf16 at [B,H,64,S]   (V transposed layout)
// mode 2: write fp32 row-major [4096][1024]  (final output)
__global__ __launch_bounds__(256) void gemm_bf16_kernel(
    const __bf16* __restrict__ A, const __bf16* __restrict__ Wt,
    const float* __restrict__ bias, void* __restrict__ outp, int mode) {
  __shared__ __bf16 As[2][128 * 40];   // 128 rows x 32 K, padded to 40
  __shared__ __bf16 Bs[2][64 * 40];    // 64 N-rows (of Wt) x 32 K

  const int tid = threadIdx.x;
  const int lane = tid & 31, w = tid >> 5;
  const int wm = w & 3, wn = w >> 2;         // wave grid 4x2
  const int lm = lane & 15, lh = lane >> 4;
  const int bm = blockIdx.x * 128, bn = blockIdx.y * 64;
  const int lr = tid >> 2, lc = (tid & 3) * 8;

  auto issue_tile = [&](int kb, int buf) {
    cp16_g2l(&A[(size_t)(bm + lr) * 1024 + kb + lc],      &As[buf][lr * 40 + lc]);
    cp16_g2l(&A[(size_t)(bm + 64 + lr) * 1024 + kb + lc], &As[buf][(lr + 64) * 40 + lc]);
    cp16_g2l(&Wt[(size_t)(bn + lr) * 1024 + kb + lc],     &Bs[buf][lr * 40 + lc]);
  };

  issue_tile(0, 0);

  v8f acc[2][2] = {};

  for (int kt = 0; kt < 32; ++kt) {
    wait_async_all();          // my async copies for tile kt are complete
    __syncthreads();           // everyone's copies done; previous tile's readers done

    const __bf16* as = As[kt & 1];
    const __bf16* bs = Bs[kt & 1];

    v16bf af[2], bfr[2];
#pragma unroll
    for (int i = 0; i < 2; ++i) {
      const __bf16* p = &as[(wm * 32 + i * 16 + lm) * 40];
      af[i] = load2x8(p + lh * 8, p + 16 + lh * 8);
    }
#pragma unroll
    for (int j = 0; j < 2; ++j) {
      const __bf16* p = &bs[(wn * 32 + j * 16 + lm) * 40];
      bfr[j] = load2x8(p + lh * 8, p + 16 + lh * 8);
    }
#pragma unroll
    for (int i = 0; i < 2; ++i)
#pragma unroll
      for (int j = 0; j < 2; ++j)
        acc[i][j] = wmma_bf16(af[i], bfr[j], acc[i][j]);

    if (kt + 1 < 32) issue_tile((kt + 1) * 32, (kt + 1) & 1);
  }

#pragma unroll
  for (int i = 0; i < 2; ++i)
#pragma unroll
    for (int j = 0; j < 2; ++j)
#pragma unroll
      for (int e = 0; e < 8; ++e) {
        int row = bm + wm * 32 + i * 16 + lh * 8 + e;   // M index (b*2048+s)
        int col = bn + wn * 32 + j * 16 + lm;           // N index (h*64+d)
        float v = acc[i][j][e] + bias[col];
        if (mode == 2) {
          ((float*)outp)[(size_t)row * 1024 + col] = v;
        } else {
          int b = row >> 11, s = row & 2047;
          int hh = col >> 6, d = col & 63;
          if (mode == 0)
            ((__bf16*)outp)[(((size_t)(b * 16 + hh)) * 2048 + s) * 64 + d] = (__bf16)v;
          else
            ((__bf16*)outp)[(((size_t)(b * 16 + hh)) * 64 + d) * 2048 + s] = (__bf16)v;
        }
      }
}

// ---------------------------------------------------------------- flash attention
// grid (S/128, H, B), 256 threads (8 waves). Wave w owns 16 q-rows.
__global__ __launch_bounds__(256) void flash_attn_kernel(
    const __bf16* __restrict__ Q, const __bf16* __restrict__ K,
    const __bf16* __restrict__ Vt, __bf16* __restrict__ O) {
  __shared__ __bf16 pbuf[8][16 * 40];   // per-wave P bounce buffer (C-layout -> A-layout)

  const int b = blockIdx.z, h = blockIdx.y;
  const int tid = threadIdx.x, lane = tid & 31, w = tid >> 5;
  const int lm = lane & 15, lh = lane >> 4;
  const size_t bh = (size_t)(b * 16 + h);
  const __bf16* Qh = Q + bh * 2048 * 64;
  const __bf16* Kh = K + bh * 2048 * 64;
  const __bf16* Vh = Vt + bh * 64 * 2048;
  const int q0 = blockIdx.x * 128 + w * 16;

  // Q A-fragments (16 rows x depth 64 -> two K=32 fragments), loaded once
  const __bf16* qrow = Qh + (size_t)(q0 + lm) * 64;
  v16bf qa0 = load2x8(qrow + lh * 8, qrow + 16 + lh * 8);
  v16bf qa1 = load2x8(qrow + 32 + lh * 8, qrow + 48 + lh * 8);

  v8f acc[4] = {};
  v8f rmax, rsum;
#pragma unroll
  for (int e = 0; e < 8; ++e) { rmax[e] = -1e30f; rsum[e] = 0.f; }

  __bf16* pb = &pbuf[w][0];

  for (int kb = 0; kb < 2048; kb += 32) {
    // prefetch next k-tile's K rows (32 lanes cover rows kb+32..kb+63)
    // and the next-tile segments of V rows
    if (kb + 32 < 2048) {
      __builtin_prefetch(Kh + (size_t)(kb + 32 + lane) * 64, 0, 3);
      __builtin_prefetch(Vh + (size_t)(lane * 2) * 2048 + kb + 32, 0, 3);
      __builtin_prefetch(Vh + (size_t)(lane * 2 + 1) * 2048 + kb + 32, 0, 3);
    }

    // ---- scores: 16 q-rows x 32 k-cols, QK^T
    v8f s0 = {}, s1 = {};
    {
      const __bf16* kr0 = Kh + (size_t)(kb + lm) * 64;        // k-cols kb..kb+15
      s0 = wmma_bf16(qa0, load2x8(kr0 + lh * 8, kr0 + 16 + lh * 8), s0);          // d 0..31
      s0 = wmma_bf16(qa1, load2x8(kr0 + 32 + lh * 8, kr0 + 48 + lh * 8), s0);     // d 32..63
      const __bf16* kr1 = Kh + (size_t)(kb + 16 + lm) * 64;    // k-cols kb+16..kb+31
      s1 = wmma_bf16(qa0, load2x8(kr1 + lh * 8, kr1 + 16 + lh * 8), s1);
      s1 = wmma_bf16(qa1, load2x8(kr1 + 32 + lh * 8, kr1 + 48 + lh * 8), s1);
    }

    // ---- online softmax (row r = lh*8+e lives across the 16 lanes of this half)
    v8f t;
#pragma unroll
    for (int e = 0; e < 8; ++e) {
      s0[e] *= 0.125f; s1[e] *= 0.125f;               // 1/sqrt(depth)
      t[e] = fmaxf(s0[e], s1[e]);
    }
#pragma unroll
    for (int m = 1; m <= 8; m <<= 1)
#pragma unroll
      for (int e = 0; e < 8; ++e) t[e] = fmaxf(t[e], __shfl_xor(t[e], m, 32));

    v8f corr, rs;
#pragma unroll
    for (int e = 0; e < 8; ++e) {
      float nm = fmaxf(rmax[e], t[e]);
      corr[e] = __expf(rmax[e] - nm);
      rmax[e] = nm;
      float p0 = __expf(s0[e] - nm), p1 = __expf(s1[e] - nm);
      s0[e] = p0; s1[e] = p1;
      rs[e] = p0 + p1;
    }
#pragma unroll
    for (int m = 1; m <= 8; m <<= 1)
#pragma unroll
      for (int e = 0; e < 8; ++e) rs[e] += __shfl_xor(rs[e], m, 32);
#pragma unroll
    for (int e = 0; e < 8; ++e) rsum[e] = rsum[e] * corr[e] + rs[e];
#pragma unroll
    for (int dt = 0; dt < 4; ++dt)
#pragma unroll
      for (int e = 0; e < 8; ++e) acc[dt][e] *= corr[e];

    // ---- P (C-layout fp32) -> LDS bf16 -> A-fragment (16x32, K=32)
#pragma unroll
    for (int e = 0; e < 8; ++e) {
      pb[(lh * 8 + e) * 40 + lm]      = (__bf16)s0[e];
      pb[(lh * 8 + e) * 40 + 16 + lm] = (__bf16)s1[e];
    }
    asm volatile("s_wait_dscnt 0" ::: "memory");
    v16bf pa = load2x8(pb + lm * 40 + lh * 8, pb + lm * 40 + 16 + lh * 8);

    // ---- out += P @ V   (V transposed: B-frag is a contiguous Vt row read)
#pragma unroll
    for (int dt = 0; dt < 4; ++dt) {
      const __bf16* vr = Vh + (size_t)(dt * 16 + lm) * 2048 + kb;
      acc[dt] = wmma_bf16(pa, load2x8(vr + lh * 8, vr + 16 + lh * 8), acc[dt]);
    }
  }

  // ---- normalize + write O[b*2048+row][h*64+d] bf16
#pragma unroll
  for (int dt = 0; dt < 4; ++dt)
#pragma unroll
    for (int e = 0; e < 8; ++e) {
      float v = acc[dt][e] / rsum[e];
      int row = q0 + lh * 8 + e;
      int col = h * 64 + dt * 16 + lm;
      O[((size_t)(b * 2048 + row)) * 1024 + col] = (__bf16)v;
    }
}

// ---------------------------------------------------------------- launch
extern "C" void kernel_launch(void* const* d_in, const int* in_sizes, int n_in,
                              void* d_out, int out_size, void* d_ws, size_t ws_size,
                              hipStream_t stream) {
  (void)in_sizes; (void)n_in; (void)out_size; (void)ws_size;
  const float* x  = (const float*)d_in[0];
  const float* wq = (const float*)d_in[1];
  const float* bq = (const float*)d_in[2];
  const float* wk = (const float*)d_in[3];
  const float* bk = (const float*)d_in[4];
  const float* wv = (const float*)d_in[5];
  const float* bv = (const float*)d_in[6];
  const float* wo = (const float*)d_in[7];
  const float* bo = (const float*)d_in[8];
  float* out = (float*)d_out;

  const size_t MB = 1024 * 1024;
  char* ws = (char*)d_ws;
  __bf16* xb  = (__bf16*)(ws + 0);        // 8 MB  x in bf16 [4096][1024]
  __bf16* wtq = (__bf16*)(ws + 8 * MB);   // 2 MB each, transposed weights
  __bf16* wtk = (__bf16*)(ws + 10 * MB);
  __bf16* wtv = (__bf16*)(ws + 12 * MB);
  __bf16* wto = (__bf16*)(ws + 14 * MB);
  __bf16* Qb  = (__bf16*)(ws + 16 * MB);  // 8 MB [B,H,S,64]
  __bf16* Kb  = (__bf16*)(ws + 24 * MB);  // 8 MB [B,H,S,64]
  __bf16* Vtb = (__bf16*)(ws + 32 * MB);  // 8 MB [B,H,64,S]
  __bf16* Ob  = (__bf16*)(ws + 40 * MB);  // 8 MB attention output [4096][1024]

  convert_x_kernel<<<4096, 256, 0, stream>>>(x, xb);
  transpose_w_kernel<<<dim3(32, 32, 4), dim3(32, 32), 0, stream>>>(
      wq, wk, wv, wo, wtq, wtk, wtv, wto);
  gemm_bf16_kernel<<<dim3(32, 16), 256, 0, stream>>>(xb, wtq, bq, Qb, 0);
  gemm_bf16_kernel<<<dim3(32, 16), 256, 0, stream>>>(xb, wtk, bk, Kb, 0);
  gemm_bf16_kernel<<<dim3(32, 16), 256, 0, stream>>>(xb, wtv, bv, Vtb, 1);
  flash_attn_kernel<<<dim3(16, 16, 2), 256, 0, stream>>>(Qb, Kb, Vtb, Ob);
  gemm_bf16_kernel<<<dim3(32, 16), 256, 0, stream>>>(Ob, wto, bo, out, 2);
}
